// DiscriminatorNetwork_14053132992646
// MI455X (gfx1250) — compile-verified, matrix operand order checked by reading, not run on hardware
//
#include <hip/hip_runtime.h>
#include <hip/hip_bf16.h>

typedef _Float16 v16h __attribute__((ext_vector_type(16)));
typedef _Float16 v8h  __attribute__((ext_vector_type(8)));
typedef float    v8f  __attribute__((ext_vector_type(8)));

#define HIDDEN 256
#define TSTEPS 512
#define GATES  1024
#define FRAGS_PER_MAT 512              /* 16 waves * 4 gate-tiles * 8 k-tiles */
#define HALFS_PER_FRAG (32*16)         /* 32 lanes * 16 halves */
#define MAT_HALFS (FRAGS_PER_MAT*HALFS_PER_FRAG) /* 262144 = 2^18 */

// Fast device transcendentals: v_exp_f32 + v_rcp_f32 only (both TRANS ops that
// co-execute with XDL WMMA), no IEEE divide expansion.
__device__ __forceinline__ float sigm(float x)
{ return __builtin_amdgcn_rcpf(1.f + __expf(-x)); }
__device__ __forceinline__ float tanh_fast(float x)
{ return 1.f - 2.f * __builtin_amdgcn_rcpf(__expf(2.f * x) + 1.f); }

// ---------------------------------------------------------------------------
// Prep kernel 1: permute + convert the 5 big weight matrices [1024x256] f32
// into f16 B-matrix fragments, per-lane contiguous, matching the CDNA5
// 16-bit WMMA B layout (lane = N col, halves hold K per the ISA striping).
// Column permutation: n = wave*64 + gate*16 + c  <->  row r = gate*256 + wave*16 + c
// so each wave owns gates i,f,g,o for a contiguous 16-unit hidden chunk.
// ---------------------------------------------------------------------------
__global__ void pack_weights(const float* __restrict__ w0, const float* __restrict__ w1,
                             const float* __restrict__ w2, const float* __restrict__ w3,
                             const float* __restrict__ w4, _Float16* __restrict__ packed)
{
    unsigned idx = blockIdx.x * blockDim.x + threadIdx.x;
    if (idx >= 5u * MAT_HALFS) return;
    unsigned m    = idx >> 18;
    unsigned rem  = idx & 0x3FFFFu;
    unsigned frag = rem >> 9;           // 0..511
    unsigned lane = (rem >> 4) & 31u;
    unsigned hh   = rem & 15u;
    unsigned kt = frag & 7u;            // k tile 0..7
    unsigned tt = (frag >> 3) & 3u;     // gate tile 0..3  (i,f,g,o)
    unsigned w  = frag >> 5;            // wave 0..15
    unsigned c  = lane & 15u;
    unsigned r  = tt * 256u + w * 16u + c;                       // source gate row
    unsigned kk = (hh < 8u ? hh : hh + 8u) + ((lane >> 4) << 3); // ISA 16-bit B K striping
    unsigned k  = kt * 32u + kk;
    const float* W = (m == 0) ? w0 : (m == 1) ? w1 : (m == 2) ? w2 : (m == 3) ? w3 : w4;
    packed[idx] = (_Float16)W[r * 256u + k];
}

// ---------------------------------------------------------------------------
// Prep kernel 2: combined bias (b_ih + b_hh), stored in the permuted column
// order so the main kernel indexes it directly per lane.
// ---------------------------------------------------------------------------
__global__ void pack_bias(const float* bi0, const float* bh0, const float* bi1, const float* bh1,
                          const float* bi2, const float* bh2, float* __restrict__ biasP)
{
    int idx = blockIdx.x * blockDim.x + threadIdx.x;
    if (idx >= 3 * 1024) return;
    int l = idx >> 10, n = idx & 1023;
    int w = n >> 6, tt = (n & 63) >> 4, c = n & 15;
    int r = tt * 256 + w * 16 + c;
    const float* bi = (l == 0) ? bi0 : (l == 1) ? bi1 : bi2;
    const float* bh = (l == 0) ? bh0 : (l == 1) ? bh1 : bh2;
    biasP[idx] = bi[r] + bh[r];
}

// ---------------------------------------------------------------------------
// Main persistent kernel: 8 blocks x 512 threads (16 waves).
// Block b owns batch rows [32b, 32b+32) = two WMMA M tiles, so every streamed
// B fragment feeds two WMMAs (2x arithmetic intensity on the L2 weight
// stream). Whole 512-step scan in one launch; wave w owns gate columns
// {i,f,g,o} x hidden[16w,16w+16); cell state lives in VGPRs.
// An opaque zero offset (refreshed per step) keeps the 2.5 MB weight stream
// inside the loop (no hoist->scratch-spill) while preserving the pointer
// provenance, so loads keep the global_load saddr+imm24 form (LOADcnt only,
// decoupled from the LDS DScnt chain).
// ---------------------------------------------------------------------------
__global__ void __launch_bounds__(512) lstm_persistent(
    const float* __restrict__ tracks,   // [256][512][2]
    const float* __restrict__ Wih0,     // [1024][2]
    const _Float16* __restrict__ packed,// 5 matrices of fragments
    const float* __restrict__ biasP,    // [3][1024] permuted
    const float* __restrict__ Wpred,    // [2][256]
    const float* __restrict__ bpred,    // [2]
    float* __restrict__ out)            // [256][2]
{
    __shared__ _Float16 hb[3][32][256]; // h state per layer, f16, batch-major
    __shared__ float    xb[32][2];      // current-step raw inputs

    const int tid   = threadIdx.x;
    const int wv    = tid >> 5;
    const int lane  = tid & 31;
    const int c16   = lane & 15;
    const int hi    = lane >> 4;
    const int bbase = blockIdx.x * 32;

    for (int i = tid; i < 3 * 32 * 256; i += 512)
        ((_Float16*)hb)[i] = (_Float16)0.f;
    __syncthreads();

    float creg[3][2][8];                // cell state: layer x M-tile x 8 rows
    #pragma unroll
    for (int l = 0; l < 3; ++l)
        #pragma unroll
        for (int mt = 0; mt < 2; ++mt)
            #pragma unroll
            for (int v = 0; v < 8; ++v) creg[l][mt][v] = 0.f;

    // Loop-invariant per-lane scalars, hoisted once for all 512 steps.
    float biasr[3][4], wih0a[4], wih0b[4];
    #pragma unroll
    for (int l = 0; l < 3; ++l)
        #pragma unroll
        for (int p = 0; p < 4; ++p)
            biasr[l][p] = biasP[l * 1024 + wv * 64 + p * 16 + c16];
    #pragma unroll
    for (int p = 0; p < 4; ++p) {
        int r = p * 256 + wv * 16 + c16;
        wih0a[p] = Wih0[r * 2 + 0];
        wih0b[p] = Wih0[r * 2 + 1];
    }

    // Opaque zero: refreshed (no-op) every time-step so weight addresses are
    // loop-variant (defeats LICM) but keep their global-pointer provenance.
    unsigned woff = 0;

    // A fragment (16x32 f16, M = batch) from an LDS h buffer, ISA striping.
    auto loadA = [&](const _Float16* base, int mt, int kt) -> v16h {
        int off = (mt * 16 + c16) * 256 + kt * 32 + hi * 8;
        v8h lo = *(const v8h*)(base + off);
        v8h hh = *(const v8h*)(base + off + 16);
        return __builtin_shufflevector(lo, hh, 0,1,2,3,4,5,6,7,8,9,10,11,12,13,14,15);
    };
    // B fragment: pre-packed, per-lane contiguous 32 bytes, streamed from L2.
    auto loadB = [&](int m, int tt, int kt) -> v16h {
        const _Float16* p =
            packed + woff
                   + (((unsigned)m * FRAGS_PER_MAT + (unsigned)((wv * 4 + tt) * 8 + kt)) * 32u
                      + (unsigned)lane) * 16u;
        v8h lo = *(const v8h*)p;
        v8h hh = *(const v8h*)(p + 8);
        return __builtin_shufflevector(lo, hh, 0,1,2,3,4,5,6,7,8,9,10,11,12,13,14,15);
    };

    auto layer = [&](int l, int mih, int mhh, const _Float16* xin) {
        __builtin_prefetch(packed + woff
                           + ((unsigned)(mhh * FRAGS_PER_MAT) + (unsigned)wv * 32u) * 512u, 0, 0);

        v8f acc[2][4];
        #pragma unroll
        for (int p = 0; p < 4; ++p) {
            float bv = biasr[l][p];
            acc[0][p] = v8f{bv, bv, bv, bv, bv, bv, bv, bv};
            acc[1][p] = acc[0][p];
        }
        if (l == 0) {
            // x @ W_ih0^T has K=2: do it with VALU from staged x.
            #pragma unroll
            for (int p = 0; p < 4; ++p)
                #pragma unroll
                for (int mt = 0; mt < 2; ++mt)
                    #pragma unroll
                    for (int v = 0; v < 8; ++v) {
                        int b = mt * 16 + v + hi * 8;
                        acc[mt][p][v] += xb[b][0] * wih0a[p] + xb[b][1] * wih0b[p];
                    }
        }
        #pragma unroll
        for (int kt = 0; kt < 8; ++kt) {
            if (xin) {
                v16h ax0 = loadA(xin, 0, kt);
                v16h ax1 = loadA(xin, 1, kt);
                v16h bf[4];
                #pragma unroll
                for (int p = 0; p < 4; ++p) bf[p] = loadB(mih, p, kt);
                #pragma unroll
                for (int p = 0; p < 4; ++p) {
                    acc[0][p] = __builtin_amdgcn_wmma_f32_16x16x32_f16(
                        false, ax0, false, bf[p], (short)0, acc[0][p], false, false);
                    acc[1][p] = __builtin_amdgcn_wmma_f32_16x16x32_f16(
                        false, ax1, false, bf[p], (short)0, acc[1][p], false, false);
                }
            }
            v16h ah0 = loadA(&hb[l][0][0], 0, kt);
            v16h ah1 = loadA(&hb[l][0][0], 1, kt);
            v16h bh[4];
            #pragma unroll
            for (int p = 0; p < 4; ++p) bh[p] = loadB(mhh, p, kt);
            #pragma unroll
            for (int p = 0; p < 4; ++p) {
                acc[0][p] = __builtin_amdgcn_wmma_f32_16x16x32_f16(
                    false, ah0, false, bh[p], (short)0, acc[0][p], false, false);
                acc[1][p] = __builtin_amdgcn_wmma_f32_16x16x32_f16(
                    false, ah1, false, bh[p], (short)0, acc[1][p], false, false);
            }
        }
        // Elementwise LSTM cell: tiles 0..3 = i,f,g,o at identical lane/VGPR slots.
        float hnew[2][8];
        #pragma unroll
        for (int mt = 0; mt < 2; ++mt)
            #pragma unroll
            for (int v = 0; v < 8; ++v) {
                float iv = sigm(acc[mt][0][v]);
                float fv = sigm(acc[mt][1][v]);
                float gv = tanh_fast(acc[mt][2][v]);
                float ov = sigm(acc[mt][3][v]);
                float cn = fv * creg[l][mt][v] + iv * gv;
                creg[l][mt][v] = cn;
                hnew[mt][v] = ov * tanh_fast(cn);
            }
        __syncthreads();                        // everyone done reading hb[l] (old)
        int j = wv * 16 + c16;
        #pragma unroll
        for (int mt = 0; mt < 2; ++mt)
            #pragma unroll
            for (int v = 0; v < 8; ++v)
                hb[l][mt * 16 + v + hi * 8][j] = (_Float16)hnew[mt][v];
        __syncthreads();                        // hb[l] (new) visible to all
    };

    for (int t = 0; t < TSTEPS; ++t) {
        // Defeat LICM of the 2.5 MB weight stream (would spill to scratch)
        // while keeping global-pointer provenance (saddr addressing form).
        asm volatile("" : "+s"(woff));

        if (tid < 64) {
            int b = tid >> 1, comp = tid & 1;
            xb[b][comp] = tracks[((bbase + b) * TSTEPS + t) * 2 + comp];
            // Speculative prefetch of next step's samples.
            __builtin_prefetch(&tracks[((bbase + b) * TSTEPS + t + 1) * 2], 0, 0);
        }
        __syncthreads();
        layer(0, -1, 0, nullptr);
        layer(1, 1, 2, &hb[0][0][0]);
        layer(2, 3, 4, &hb[1][0][0]);
    }

    // Epilogue: out = elu(h2) @ W_pred^T + b_pred  (tiny: 2 outputs per row)
    if (tid < 64) {
        int b = tid >> 1, o = tid & 1;
        float s = bpred[o];
        for (int jj = 0; jj < 256; ++jj) {
            float hv = (float)hb[2][b][jj];
            float e = hv > 0.f ? hv : (__expf(hv) - 1.f);
            s += e * Wpred[o * 256 + jj];
        }
        out[(bbase + b) * 2 + o] = s;
    }
}

extern "C" void kernel_launch(void* const* d_in, const int* in_sizes, int n_in,
                              void* d_out, int out_size, void* d_ws, size_t ws_size,
                              hipStream_t stream)
{
    const float* tracks = (const float*)d_in[0];
    const float* Wih0   = (const float*)d_in[1];
    const float* Whh0   = (const float*)d_in[2];
    const float* bih0   = (const float*)d_in[3];
    const float* bhh0   = (const float*)d_in[4];
    const float* Wih1   = (const float*)d_in[5];
    const float* Whh1   = (const float*)d_in[6];
    const float* bih1   = (const float*)d_in[7];
    const float* bhh1   = (const float*)d_in[8];
    const float* Wih2   = (const float*)d_in[9];
    const float* Whh2   = (const float*)d_in[10];
    const float* bih2   = (const float*)d_in[11];
    const float* bhh2   = (const float*)d_in[12];
    const float* Wpred  = (const float*)d_in[13];
    const float* bpred  = (const float*)d_in[14];

    _Float16* packed = (_Float16*)d_ws;                                  // 2.5 MB
    float*    biasP  = (float*)((char*)d_ws + (size_t)5 * MAT_HALFS * sizeof(_Float16));

    pack_weights<<<(5 * MAT_HALFS + 255) / 256, 256, 0, stream>>>(
        Whh0, Wih1, Whh1, Wih2, Whh2, packed);
    pack_bias<<<(3 * 1024 + 255) / 256, 256, 0, stream>>>(
        bih0, bhh0, bih1, bhh1, bih2, bhh2, biasP);
    lstm_persistent<<<8, 512, 0, stream>>>(
        tracks, Wih0, packed, biasP, Wpred, bpred, (float*)d_out);
}
